// _GraphConvolutionLayer_23742579212572
// MI455X (gfx1250) — matrix operand order repaired and checked by reading, not compiled, not run on hardware
//
#include <hip/hip_runtime.h>

// ---------------------------------------------------------------------------
// CDNA5 (gfx1250) graph-conv layer: bf16 WMMA GEMM pipeline, fp32 accumulate.
// All GEMMs are "TN": C[m,n] = sum_k A[m,k]*B[n,k], both operands K-contiguous
// bf16 (stored as u16), accumulated with v_wmma_f32_16x16x32_bf16.
// ---------------------------------------------------------------------------

typedef __attribute__((ext_vector_type(16))) __bf16          v16bf;
typedef __attribute__((ext_vector_type(8)))  __bf16          v8bf;
typedef __attribute__((ext_vector_type(8)))  float           v8f;
typedef __attribute__((ext_vector_type(8)))  unsigned short  v8us;

union FragA { v16bf v; v8bf h[2]; };

__device__ __forceinline__ unsigned short f2bf(float f) {
  union { float f; unsigned int u; } x; x.f = f;
  unsigned int u = x.u;
  u += 0x7fffu + ((u >> 16) & 1u);           // round-to-nearest-even
  return (unsigned short)(u >> 16);
}

// ---------------------------------------------------------------------------
// Pack: dst_bf16[i] = src_f32[i*stride + off]   (stride=2 extracts rel channel)
// ---------------------------------------------------------------------------
__global__ void pack_strided_kernel(unsigned short* __restrict__ dst,
                                    const float* __restrict__ src,
                                    long n, int stride, int off) {
  long i = (long)blockIdx.x * blockDim.x + threadIdx.x;
  if (i < n) dst[i] = f2bf(src[i * (long)stride + off]);
}

// ---------------------------------------------------------------------------
// Transpose-pack: dst_bf16[c*R + r] = src_f32[r*rowStride + c*colStride + off]
// 32x32 LDS tile; both global accesses coalesced. Dims are multiples of 32.
// ---------------------------------------------------------------------------
__global__ void transpose_pack_kernel(unsigned short* __restrict__ dst,
                                      const float* __restrict__ src,
                                      int R, long rowStride, int colStride, int off) {
  __shared__ float tile[32][33];
  const int c0 = blockIdx.x * 32, r0 = blockIdx.y * 32;
#pragma unroll
  for (int j = 0; j < 32; j += 8) {
    long r = r0 + threadIdx.y + j;
    long c = c0 + threadIdx.x;
    tile[threadIdx.y + j][threadIdx.x] = src[r * rowStride + c * (long)colStride + off];
  }
  __syncthreads();
#pragma unroll
  for (int j = 0; j < 32; j += 8) {
    long c = c0 + threadIdx.y + j;
    long r = r0 + threadIdx.x;
    dst[c * (long)R + r] = f2bf(tile[threadIdx.x][threadIdx.y + j]);
  }
}

// ---------------------------------------------------------------------------
// Normalizers: dst[r] = 1/(row_sum + eps)  and  dst[c] = 1/(col_sum + eps)
// ---------------------------------------------------------------------------
__global__ void rowsum_inv_kernel(float* __restrict__ dst, const float* __restrict__ src,
                                  int C, long rowStride, int colStride, int off) {
  __shared__ float sm[256];
  const long r = blockIdx.x;
  const float* p = src + r * rowStride + off;
  float s = 0.f;
  for (int c = threadIdx.x; c < C; c += 256) s += p[(long)c * colStride];
  sm[threadIdx.x] = s;
  __syncthreads();
  for (int w = 128; w > 0; w >>= 1) {
    if ((int)threadIdx.x < w) sm[threadIdx.x] += sm[threadIdx.x + w];
    __syncthreads();
  }
  if (threadIdx.x == 0) dst[r] = 1.f / (sm[0] + 1e-7f);
}

__global__ void colsum_inv_kernel(float* __restrict__ dst, const float* __restrict__ src,
                                  int R, long rowStride, int colStride, int off) {
  long c = (long)blockIdx.x * blockDim.x + threadIdx.x;
  const float* p = src + c * (long)colStride + off;
  float s = 0.f;
  for (int r = 0; r < R; ++r) s += p[(long)r * rowStride];
  dst[c] = 1.f / (s + 1e-7f);
}

// ---------------------------------------------------------------------------
// WMMA GEMM: C[m,n] = sum_k A[m,k] * B[n,k]   (A:[M,K] bf16, B:[N,K] bf16)
// Block = 256 threads (8 wave32), WG tile 128x128, wave tile 32x64 (2x4 accums).
// mode 0: outF[m*N+n]  = residual[m*N+n] + scale*rowinv[m]*acc
// mode 1: outF[m*N+n] += scale*rowinv[m]*acc
// mode 2: outT[n*M+m]  = bf16(relu(acc + bias[n]))      (transposed FC output)
// M,N multiples of 128; K multiple of 32.
// ---------------------------------------------------------------------------
__global__ __launch_bounds__(256)
void wmma_gemm_kernel(const unsigned short* __restrict__ A,
                      const unsigned short* __restrict__ B,
                      int M, int N, int K,
                      const float* __restrict__ bias,
                      const float* __restrict__ rowinv,
                      const float* __restrict__ residual,
                      float* __restrict__ outF,
                      unsigned short* __restrict__ outT,
                      float scale, int mode) {
  const int lane  = threadIdx.x & 31;
  const int wave  = threadIdx.x >> 5;
  const int waveM = wave & 3;        // 4 waves along M
  const int waveN = wave >> 2;       // 2 waves along N
  const int half  = lane >> 4;       // K-half selector (contiguous k order)
  const int lrow  = lane & 15;       // row (A) / col (B) within 16

  const int mW = blockIdx.x * 128 + waveM * 32;
  const int nW = blockIdx.y * 128 + waveN * 64;

  v8f zero;
#pragma unroll
  for (int i = 0; i < 8; ++i) zero[i] = 0.f;
  v8f acc[2][4];
#pragma unroll
  for (int mt = 0; mt < 2; ++mt)
#pragma unroll
    for (int nt = 0; nt < 4; ++nt) acc[mt][nt] = zero;

  // Per-lane fragment bases (contiguous k order, k = 16*half + s, same for A and B):
  // A 16x32 bf16: lanes 0-15 hold K = k0..k0+15 (row = lane),
  //               lanes 16-31 hold K = k0+16..31.
  // B 32x16 bf16: lane n (0-15) holds K = k0..k0+15; lane n+16 K = k0+16..31.
  const unsigned short* aBase = A + (size_t)(mW + lrow) * K + half * 16;
  const unsigned short* bBase = B + (size_t)(nW + lrow) * K + half * 16;

  for (int k0 = 0; k0 < K; k0 += 32) {
    FragA a[2];
    v16bf b[4];
#pragma unroll
    for (int mt = 0; mt < 2; ++mt) {
      const unsigned short* p = aBase + (size_t)mt * 16 * K + k0;
      a[mt].h[0] = *(const v8bf*)(p);
      a[mt].h[1] = *(const v8bf*)(p + 8);
    }
#pragma unroll
    for (int nt = 0; nt < 4; ++nt) {
      const unsigned short* p = bBase + (size_t)nt * 16 * K + k0;
      b[nt] = *(const v16bf*)(p);
    }
    if (k0 + 256 < K) {   // prefetch the streamed A operand (global_prefetch_b8)
      __builtin_prefetch(aBase + k0 + 256, 0, 1);
      __builtin_prefetch(aBase + (size_t)16 * K + k0 + 256, 0, 1);
    }
#pragma unroll
    for (int mt = 0; mt < 2; ++mt)
#pragma unroll
      for (int nt = 0; nt < 4; ++nt)
        acc[mt][nt] = __builtin_amdgcn_wmma_f32_16x16x32_bf16(
            false, a[mt].v, false, b[nt], (short)0, acc[mt][nt], false, false);
  }

  // C/D layout: lanes 0-15: elem i -> (M = i,   N = lane);
  //             lanes 16-31: elem i -> (M = 8+i, N = lane-16).
  if (mode == 2) {
#pragma unroll
    for (int mt = 0; mt < 2; ++mt) {
#pragma unroll
      for (int nt = 0; nt < 4; ++nt) {
        const int n  = nW + nt * 16 + lrow;
        const float bb = bias[n];
        const int m0 = mW + mt * 16 + half * 8;
        v8us ob;
#pragma unroll
        for (int i = 0; i < 8; ++i) {
          float v = acc[mt][nt][i] + bb;
          v = v > 0.f ? v : 0.f;
          ob[i] = f2bf(v);
        }
        *(v8us*)(outT + (size_t)n * M + m0) = ob;
      }
    }
  } else {
#pragma unroll
    for (int mt = 0; mt < 2; ++mt) {
#pragma unroll
      for (int nt = 0; nt < 4; ++nt) {
        const int n  = nW + nt * 16 + lrow;
        const int m0 = mW + mt * 16 + half * 8;
#pragma unroll
        for (int i = 0; i < 8; ++i) {
          const int m = m0 + i;
          const float v = acc[mt][nt][i] * scale * rowinv[m];
          const size_t idx = (size_t)m * N + n;
          if (mode == 0) outF[idx] = residual[idx] + v;
          else           outF[idx] += v;
        }
      }
    }
  }
}

// ---------------------------------------------------------------------------
// Host orchestration
// ---------------------------------------------------------------------------
extern "C" void kernel_launch(void* const* d_in, const int* in_sizes, int n_in,
                              void* d_out, int out_size, void* d_ws, size_t ws_size,
                              hipStream_t stream) {
  (void)in_sizes; (void)n_in; (void)out_size; (void)ws_size;
  const int D = 512, No = 2048, Na = 2048, Nr = 8192;

  const float* feat_obj = (const float*)d_in[0];
  const float* feat_att = (const float*)d_in[1];
  const float* feat_rel = (const float*)d_in[2];
  const float* map_att  = (const float*)d_in[3];   // [No, Na]
  const float* map_oo   = (const float*)d_in[4];   // [No, No]
  const float* map_rel  = (const float*)d_in[5];   // [No, Nr, 2]
  const float* W0 = (const float*)d_in[6];
  const float* b0 = (const float*)d_in[7];
  const float* W1 = (const float*)d_in[8];
  const float* b1 = (const float*)d_in[9];
  const float* W2 = (const float*)d_in[10];
  const float* b2 = (const float*)d_in[11];

  // Workspace carve-out (~178 MB), 256 B aligned slices.
  char* w = (char*)d_ws;
  auto alloc = [&](size_t bytes) -> char* {
    char* p = w; w += (bytes + 255) & ~(size_t)255; return p;
  };
  unsigned short* bfFO  = (unsigned short*)alloc((size_t)No * D * 2);
  unsigned short* bfFA  = (unsigned short*)alloc((size_t)Na * D * 2);
  unsigned short* bfFR  = (unsigned short*)alloc((size_t)Nr * D * 2);
  unsigned short* bfW0  = (unsigned short*)alloc((size_t)D * D * 2);
  unsigned short* bfW1  = (unsigned short*)alloc((size_t)D * D * 2);
  unsigned short* bfW2  = (unsigned short*)alloc((size_t)D * D * 2);
  unsigned short* fc0T  = (unsigned short*)alloc((size_t)D * No * 2);  // [D, No]
  unsigned short* fc1T  = (unsigned short*)alloc((size_t)D * Na * 2);  // [D, Na]
  unsigned short* fc2T  = (unsigned short*)alloc((size_t)D * Nr * 2);  // [D, Nr]
  unsigned short* bfATT  = (unsigned short*)alloc((size_t)No * Na * 2); // map_att
  unsigned short* bfATTT = (unsigned short*)alloc((size_t)Na * No * 2); // map_att^T
  unsigned short* bfOO   = (unsigned short*)alloc((size_t)No * No * 2); // map_oo
  unsigned short* bfS    = (unsigned short*)alloc((size_t)No * Nr * 2); // ch0
  unsigned short* bfO    = (unsigned short*)alloc((size_t)No * Nr * 2); // ch1
  unsigned short* bfST   = (unsigned short*)alloc((size_t)Nr * No * 2); // ch0^T
  unsigned short* bfOT   = (unsigned short*)alloc((size_t)Nr * No * 2); // ch1^T
  float* invCA  = (float*)alloc((size_t)Na * 4);   // colsum(map_att)
  float* invRA  = (float*)alloc((size_t)No * 4);   // rowsum(map_att)
  float* invROO = (float*)alloc((size_t)No * 4);   // rowsum(map_oo)
  float* invRS  = (float*)alloc((size_t)No * 4);   // rowsum(ch0)
  float* invRO  = (float*)alloc((size_t)No * 4);   // rowsum(ch1)
  float* invCS  = (float*)alloc((size_t)Nr * 4);   // colsum(ch0)
  float* invCO  = (float*)alloc((size_t)Nr * 4);   // colsum(ch1)

  float* out_obj = (float*)d_out;                       // feat_obj_updated
  float* out_att = out_obj + (size_t)No * D;            // feat_att_updated
  float* out_rel = out_att + (size_t)Na * D;            // feat_rel_updated

  auto pack = [&](unsigned short* dst, const float* src, long n, int stride, int off) {
    pack_strided_kernel<<<dim3((unsigned)((n + 255) / 256)), dim3(256), 0, stream>>>(
        dst, src, n, stride, off);
  };

  // --- bf16 packs -----------------------------------------------------------
  pack(bfFO, feat_obj, (long)No * D, 1, 0);
  pack(bfFA, feat_att, (long)Na * D, 1, 0);
  pack(bfFR, feat_rel, (long)Nr * D, 1, 0);
  pack(bfW0, W0, (long)D * D, 1, 0);
  pack(bfW1, W1, (long)D * D, 1, 0);
  pack(bfW2, W2, (long)D * D, 1, 0);
  pack(bfATT, map_att, (long)No * Na, 1, 0);
  pack(bfOO,  map_oo,  (long)No * No, 1, 0);
  pack(bfS,   map_rel, (long)No * Nr, 2, 0);
  pack(bfO,   map_rel, (long)No * Nr, 2, 1);

  dim3 tb(32, 8);
  transpose_pack_kernel<<<dim3(Na / 32, No / 32), tb, 0, stream>>>(
      bfATTT, map_att, No, (long)Na, 1, 0);
  transpose_pack_kernel<<<dim3(Nr / 32, No / 32), tb, 0, stream>>>(
      bfST, map_rel, No, (long)2 * Nr, 2, 0);
  transpose_pack_kernel<<<dim3(Nr / 32, No / 32), tb, 0, stream>>>(
      bfOT, map_rel, No, (long)2 * Nr, 2, 1);

  // --- normalizers ----------------------------------------------------------
  rowsum_inv_kernel<<<No, 256, 0, stream>>>(invRA,  map_att, Na, (long)Na, 1, 0);
  rowsum_inv_kernel<<<No, 256, 0, stream>>>(invROO, map_oo,  No, (long)No, 1, 0);
  rowsum_inv_kernel<<<No, 256, 0, stream>>>(invRS,  map_rel, Nr, (long)2 * Nr, 2, 0);
  rowsum_inv_kernel<<<No, 256, 0, stream>>>(invRO,  map_rel, Nr, (long)2 * Nr, 2, 1);
  colsum_inv_kernel<<<Na / 256, 256, 0, stream>>>(invCA, map_att, No, (long)Na, 1, 0);
  colsum_inv_kernel<<<Nr / 256, 256, 0, stream>>>(invCS, map_rel, No, (long)2 * Nr, 2, 0);
  colsum_inv_kernel<<<Nr / 256, 256, 0, stream>>>(invCO, map_rel, No, (long)2 * Nr, 2, 1);

  // --- FC GEMMs: fcT = bf16(relu(X @ W^T + b))^T  (mode 2) ------------------
  wmma_gemm_kernel<<<dim3(No / 128, D / 128), 256, 0, stream>>>(
      bfFO, bfW0, No, D, D, b0, nullptr, nullptr, nullptr, fc0T, 1.f, 2);
  wmma_gemm_kernel<<<dim3(Na / 128, D / 128), 256, 0, stream>>>(
      bfFA, bfW1, Na, D, D, b1, nullptr, nullptr, nullptr, fc1T, 1.f, 2);
  wmma_gemm_kernel<<<dim3(Nr / 128, D / 128), 256, 0, stream>>>(
      bfFR, bfW2, Nr, D, D, b2, nullptr, nullptr, nullptr, fc2T, 1.f, 2);

  // --- Aggregations with fused normalization / scale / residual -------------
  // feat_att_updated = feat_att + (map_att^T @ fc0) * invCA
  wmma_gemm_kernel<<<dim3(Na / 128, D / 128), 256, 0, stream>>>(
      bfATTT, fc0T, Na, D, No, nullptr, invCA, feat_att, out_att, nullptr, 1.0f, 0);
  // feat_obj_updated = feat_obj + ( s_att + s_obj + s_rel_sub + s_rel_obj ) / 4
  wmma_gemm_kernel<<<dim3(No / 128, D / 128), 256, 0, stream>>>(
      bfATT, fc1T, No, D, Na, nullptr, invRA, feat_obj, out_obj, nullptr, 0.25f, 0);
  wmma_gemm_kernel<<<dim3(No / 128, D / 128), 256, 0, stream>>>(
      bfOO, fc0T, No, D, No, nullptr, invROO, nullptr, out_obj, nullptr, 0.25f, 1);
  wmma_gemm_kernel<<<dim3(No / 128, D / 128), 256, 0, stream>>>(
      bfS, fc2T, No, D, Nr, nullptr, invRS, nullptr, out_obj, nullptr, 0.25f, 1);
  wmma_gemm_kernel<<<dim3(No / 128, D / 128), 256, 0, stream>>>(
      bfO, fc2T, No, D, Nr, nullptr, invRO, nullptr, out_obj, nullptr, 0.25f, 1);
  // feat_rel_updated = feat_rel + ( s_obj_sub + s_obj_obj ) / 2
  wmma_gemm_kernel<<<dim3(Nr / 128, D / 128), 256, 0, stream>>>(
      bfST, fc0T, Nr, D, No, nullptr, invCS, feat_rel, out_rel, nullptr, 0.5f, 0);
  wmma_gemm_kernel<<<dim3(Nr / 128, D / 128), 256, 0, stream>>>(
      bfOT, fc0T, Nr, D, No, nullptr, invCO, nullptr, out_rel, nullptr, 0.5f, 1);
}
